// MyFormer_22179211116673
// MI455X (gfx1250) — matrix operand order repaired and checked by reading, not compiled, not run on hardware
//
#include <hip/hip_runtime.h>
#include <hip/hip_bf16.h>
#include <cmath>

typedef __bf16 bf16_t;
typedef bf16_t v16bf __attribute__((ext_vector_type(16)));
typedef bf16_t v8bf  __attribute__((ext_vector_type(8)));
typedef float  v8f   __attribute__((ext_vector_type(8)));

#define DEVINL __device__ __forceinline__

DEVINL bf16_t f2bf(float f) {
  unsigned u = __builtin_bit_cast(unsigned, f);
  unsigned r = u + 0x7FFFu + ((u >> 16) & 1u);
  unsigned short h = (unsigned short)(r >> 16);
  return __builtin_bit_cast(bf16_t, h);
}

DEVINL float waveSum(float v) {
#pragma unroll
  for (int m = 16; m; m >>= 1) v += __shfl_xor(v, m, 32);
  return v;
}
DEVINL float waveMax(float v) {
#pragma unroll
  for (int m = 16; m; m >>= 1) v = fmaxf(v, __shfl_xor(v, m, 32));
  return v;
}
DEVINL float blockSum(float v, float* sm) {
  int lane = threadIdx.x & 31, w = threadIdx.x >> 5;
  v = waveSum(v);
  if (lane == 0) sm[w] = v;
  __syncthreads();
  float r = (threadIdx.x < (blockDim.x >> 5)) ? sm[threadIdx.x] : 0.f;
  if (w == 0) { r = waveSum(r); if (lane == 0) sm[0] = r; }
  __syncthreads();
  float out = sm[0];
  __syncthreads();
  return out;
}
DEVINL float blockMax(float v, float* sm) {
  int lane = threadIdx.x & 31, w = threadIdx.x >> 5;
  v = waveMax(v);
  if (lane == 0) sm[w] = v;
  __syncthreads();
  float r = (threadIdx.x < (blockDim.x >> 5)) ? sm[threadIdx.x] : -3.4e38f;
  if (w == 0) { r = waveMax(r); if (lane == 0) sm[0] = r; }
  __syncthreads();
  float out = sm[0];
  __syncthreads();
  return out;
}

// ---------------------------------------------------------------------------
// Generic batched GEMM:  C[b] = A[b] (MxK, row-major) * Bt[b]^T (Bt is NxK
// row-major) + bias + resid.  Optional f32 and/or bf16 outputs, each either
// normal [M,N] or transposed [N,M] layout.  One wave per 16x16 tile,
// v_wmma_f32_16x16x32_bf16, all fragment loads are contiguous b128.
// ---------------------------------------------------------------------------
struct GemmP {
  const bf16_t* A;  long aB;  int lda;
  const bf16_t* Bt; long bB;  int ldb;
  const float* bias;
  const float* resid; long rB;
  float*  Cf; long cfB; int cfTrans; int ldcf;
  bf16_t* Cb; long cbB; int cbTrans; int ldcb;
  int M, N, K;
};

__global__ __launch_bounds__(256) void gemm_bf16_wmma(GemmP p) {
  const int wave = threadIdx.x >> 5;
  const int lane = threadIdx.x & 31;
  const int tilesN = p.N >> 4;
  const long tile = (long)blockIdx.x * 8 + wave;
  if (tile >= (long)(p.M >> 4) * tilesN) return;
  const int tm = (int)(tile / tilesN), tn = (int)(tile % tilesN);
  const int b = blockIdx.y;
  const int mr = lane & 15, half = lane >> 4;
  const bf16_t* aP = p.A + (long)b * p.aB + (long)(tm * 16 + mr) * p.lda + half * 8;
  const bf16_t* bP = p.Bt + (long)b * p.bB + (long)(tn * 16 + mr) * p.ldb + half * 8;
  v8f acc = {0.f, 0.f, 0.f, 0.f, 0.f, 0.f, 0.f, 0.f};
  for (int k = 0; k < p.K; k += 32) {
    union { v16bf v; v8bf h[2]; } ua, ub;
    ua.h[0] = *(const v8bf*)(aP + k);
    ua.h[1] = *(const v8bf*)(aP + k + 16);
    ub.h[0] = *(const v8bf*)(bP + k);
    ub.h[1] = *(const v8bf*)(bP + k + 16);
    acc = __builtin_amdgcn_wmma_f32_16x16x32_bf16(false, ua.v, false, ub.v,
                                                  (short)0, acc, false, false);
  }
  const int co = tn * 16 + mr;
  const int m0 = tm * 16 + half * 8;
  float add = p.bias ? p.bias[co] : 0.f;
  float vals[8];
#pragma unroll
  for (int r = 0; r < 8; r++) {
    float v = acc[r] + add;
    if (p.resid) v += p.resid[(long)b * p.rB + (long)(m0 + r) * p.N + co];
    vals[r] = v;
  }
  if (p.Cf) {
    float* base = p.Cf + (long)b * p.cfB;
    if (p.cfTrans) {
      v8f vv;
#pragma unroll
      for (int r = 0; r < 8; r++) vv[r] = vals[r];
      *(v8f*)(base + (long)co * p.ldcf + m0) = vv;
    } else {
#pragma unroll
      for (int r = 0; r < 8; r++) base[(long)(m0 + r) * p.ldcf + co] = vals[r];
    }
  }
  if (p.Cb) {
    bf16_t* base = p.Cb + (long)b * p.cbB;
    if (p.cbTrans) {
      v8bf vv;
#pragma unroll
      for (int r = 0; r < 8; r++) vv[r] = f2bf(vals[r]);
      *(v8bf*)(base + (long)co * p.ldcb + m0) = vv;
    } else {
#pragma unroll
      for (int r = 0; r < 8; r++) base[(long)(m0 + r) * p.ldcb + co] = f2bf(vals[r]);
    }
  }
}

// ---------------------------------------------------------------------------
// Elementwise / reduction kernels
// ---------------------------------------------------------------------------
__global__ void cvt_f32_bf16(const float* __restrict__ in, bf16_t* __restrict__ out, long n) {
  long i = (long)blockIdx.x * blockDim.x + threadIdx.x;
  if (i < n) out[i] = f2bf(in[i]);
}

// LayerNorm over 256 channels, one wave per row, bf16 output.
__global__ __launch_bounds__(256) void ln_rows256(const float* __restrict__ x,
    const float* __restrict__ g, const float* __restrict__ bta,
    bf16_t* __restrict__ out, long rows) {
  int wave = threadIdx.x >> 5, lane = threadIdx.x & 31;
  long row = (long)blockIdx.x * 8 + wave;
  if (row >= rows) return;
  const float* xr = x + row * 256;
  float v[8], s = 0.f, s2 = 0.f;
#pragma unroll
  for (int j = 0; j < 8; j++) { v[j] = xr[lane + 32 * j]; s += v[j]; s2 += v[j] * v[j]; }
  s = waveSum(s); s2 = waveSum(s2);
  float mean = s * (1.f / 256.f);
  float var = s2 * (1.f / 256.f) - mean * mean;
  float rstd = rsqrtf(var + 1e-5f);
  bf16_t* o = out + row * 256;
#pragma unroll
  for (int j = 0; j < 8; j++) {
    int c = lane + 32 * j;
    o[c] = f2bf((v[j] - mean) * rstd * g[c] + bta[c]);
  }
}

// Softmax over 256 channels (eff-attn q), one wave per row.
__global__ __launch_bounds__(256) void softmax_rows256(const float* __restrict__ x,
    bf16_t* __restrict__ out, long rows) {
  int wave = threadIdx.x >> 5, lane = threadIdx.x & 31;
  long row = (long)blockIdx.x * 8 + wave;
  if (row >= rows) return;
  const float* xr = x + row * 256;
  float v[8], mx = -3.4e38f;
#pragma unroll
  for (int j = 0; j < 8; j++) { v[j] = xr[lane + 32 * j]; mx = fmaxf(mx, v[j]); }
  mx = waveMax(mx);
  float s = 0.f;
#pragma unroll
  for (int j = 0; j < 8; j++) { v[j] = expf(v[j] - mx); s += v[j]; }
  s = waveSum(s);
  float inv = 1.f / s;
  bf16_t* o = out + row * 256;
#pragma unroll
  for (int j = 0; j < 8; j++) o[lane + 32 * j] = f2bf(v[j] * inv);
}

// Softmax over a length-`len` contiguous row (eff-attn k, transposed layout).
__global__ __launch_bounds__(256) void softmax_len(const float* __restrict__ in,
    bf16_t* __restrict__ out, int len) {
  __shared__ float sm[8];
  long row = blockIdx.x;
  const float* src = in + row * (long)len;
  bf16_t* dst = out + row * (long)len;
  float mx = -3.4e38f;
  for (int j = threadIdx.x; j < len; j += 256) mx = fmaxf(mx, src[j]);
  mx = blockMax(mx, sm);
  float s = 0.f;
  for (int j = threadIdx.x; j < len; j += 256) s += expf(src[j] - mx);
  s = blockSum(s, sm);
  float inv = 1.f / s;
  for (int j = threadIdx.x; j < len; j += 256) dst[j] = f2bf(expf(src[j] - mx) * inv);
}

// L2-normalize q,k rows of qkvT [B,768,3136] (rows 0..511 per batch) -> bf16.
__global__ __launch_bounds__(256) void l2norm_rows(const float* __restrict__ qkvT,
    bf16_t* __restrict__ out) {
  __shared__ float sm[8];
  int b = blockIdx.x >> 9, rr = blockIdx.x & 511;
  const float* src = qkvT + ((long)b * 768 + rr) * 3136;
  bf16_t* dst = out + ((long)b * 512 + rr) * 3136;
  float ss = 0.f;
  for (int j = threadIdx.x; j < 3136; j += 256) { float v = src[j]; ss += v * v; }
  ss = blockSum(ss, sm);
  float scale = 1.f / fmaxf(sqrtf(ss), 1e-12f);
  for (int j = threadIdx.x; j < 3136; j += 256) dst[j] = f2bf(src[j] * scale);
}

// Depthwise 3x3 conv (+bias) + skip(f) + LayerNorm(1024) + exact GELU -> bf16.
__global__ __launch_bounds__(256) void dwconv_ln_gelu(const float* __restrict__ f,
    const float* __restrict__ w, const float* __restrict__ wb,
    const float* __restrict__ lg, const float* __restrict__ lb,
    bf16_t* __restrict__ out) {
  __shared__ float sm[8];
  int n = blockIdx.x, b = blockIdx.y;
  int py = n / 56, px = n - py * 56;
  const float* fb = f + ((long)b * 3136 + n) * 1024;
  float d[4];
  float s = 0.f, s2 = 0.f;
#pragma unroll
  for (int j = 0; j < 4; j++) {
    int c = threadIdx.x + 256 * j;
    float acc = 0.f;
#pragma unroll
    for (int dy = -1; dy <= 1; dy++) {
      int yy = py + dy; if (yy < 0 || yy >= 56) continue;
#pragma unroll
      for (int dx = -1; dx <= 1; dx++) {
        int xx = px + dx; if (xx < 0 || xx >= 56) continue;
        acc += f[((long)b * 3136 + yy * 56 + xx) * 1024 + c] * w[c * 9 + (dy + 1) * 3 + (dx + 1)];
      }
    }
    float val = acc + wb[c] + fb[c];
    d[j] = val; s += val; s2 += val * val;
  }
  s = blockSum(s, sm); s2 = blockSum(s2, sm);
  float mean = s * (1.f / 1024.f);
  float var = s2 * (1.f / 1024.f) - mean * mean;
  float rstd = rsqrtf(var + 1e-5f);
  bf16_t* o = out + ((long)b * 3136 + n) * 1024;
#pragma unroll
  for (int j = 0; j < 4; j++) {
    int c = threadIdx.x + 256 * j;
    float y = (d[j] - mean) * rstd * lg[c] + lb[c];
    float gl = 0.5f * y * (1.f + erff(y * 0.70710678118654752f));
    o[c] = f2bf(gl);
  }
}

// Channel-attn scores: per (b,h) S = q k^T over N=3136 (WMMA), *temp, softmax.
__global__ __launch_bounds__(128) void chan_attn_scores(const bf16_t* __restrict__ qk,
    const float* __restrict__ temp, bf16_t* __restrict__ attn) {
  __shared__ float S[32][33];
  int h = blockIdx.x, b = blockIdx.y;
  int wave = threadIdx.x >> 5, lane = threadIdx.x & 31;
  int tm = wave >> 1, tn = wave & 1;
  int mr = lane & 15, half = lane >> 4;
  const bf16_t* aP = qk + ((long)(b * 512 + h * 32 + tm * 16 + mr)) * 3136 + half * 8;
  const bf16_t* bP = qk + ((long)(b * 512 + 256 + h * 32 + tn * 16 + mr)) * 3136 + half * 8;
  v8f acc = {0.f, 0.f, 0.f, 0.f, 0.f, 0.f, 0.f, 0.f};
  for (int k = 0; k < 3136; k += 32) {
    union { v16bf v; v8bf hh[2]; } ua, ub;
    ua.hh[0] = *(const v8bf*)(aP + k);
    ua.hh[1] = *(const v8bf*)(aP + k + 16);
    ub.hh[0] = *(const v8bf*)(bP + k);
    ub.hh[1] = *(const v8bf*)(bP + k + 16);
    acc = __builtin_amdgcn_wmma_f32_16x16x32_bf16(false, ua.v, false, ub.v,
                                                  (short)0, acc, false, false);
  }
  int co = tn * 16 + mr, m0 = tm * 16 + half * 8;
#pragma unroll
  for (int r = 0; r < 8; r++) S[m0 + r][co] = acc[r];
  __syncthreads();
  if (threadIdx.x < 32) {
    int row = threadIdx.x;
    float t = temp[h];
    float mx = -3.4e38f;
#pragma unroll
    for (int e = 0; e < 32; e++) mx = fmaxf(mx, S[row][e] * t);
    float s = 0.f;
#pragma unroll
    for (int e = 0; e < 32; e++) s += expf(S[row][e] * t - mx);
    float inv = 1.f / s;
    bf16_t* dst = attn + ((long)((b * 8 + h) * 32 + row)) * 32;
#pragma unroll
    for (int e = 0; e < 32; e++) dst[e] = f2bf(expf(S[row][e] * t - mx) * inv);
  }
}

// Channel-attn output: out[b,n,h*32+dd] = sum_e v[b,n,512+h*32+e] * attn[b,h,dd,e]
__global__ __launch_bounds__(256) void chan_attn_out(const bf16_t* __restrict__ qkv_nb,
    const bf16_t* __restrict__ attn, bf16_t* __restrict__ out) {
  int wave = threadIdx.x >> 5, lane = threadIdx.x & 31;
  int tile = blockIdx.x * 8 + wave;   // 0..391
  int h = blockIdx.y, b = blockIdx.z;
  int tm = tile >> 1, tn = tile & 1;
  int mr = lane & 15, half = lane >> 4;
  const bf16_t* aP = qkv_nb + ((long)(b * 3136 + tm * 16 + mr)) * 768 + 512 + h * 32 + half * 8;
  const bf16_t* bP = attn + ((long)((b * 8 + h) * 32 + tn * 16 + mr)) * 32 + half * 8;
  union { v16bf v; v8bf hh[2]; } ua, ub;
  ua.hh[0] = *(const v8bf*)(aP);
  ua.hh[1] = *(const v8bf*)(aP + 16);
  ub.hh[0] = *(const v8bf*)(bP);
  ub.hh[1] = *(const v8bf*)(bP + 16);
  v8f acc = {0.f, 0.f, 0.f, 0.f, 0.f, 0.f, 0.f, 0.f};
  acc = __builtin_amdgcn_wmma_f32_16x16x32_bf16(false, ua.v, false, ub.v,
                                                (short)0, acc, false, false);
  int m0 = tm * 16 + half * 8;
  int c = h * 32 + tn * 16 + mr;
#pragma unroll
  for (int r = 0; r < 8; r++) out[((long)(b * 3136) + m0 + r) * 256 + c] = f2bf(acc[r]);
}

// ---------------------------------------------------------------------------
static inline void run_gemm(hipStream_t s, const GemmP& p, int batches) {
  int tiles = (p.M / 16) * (p.N / 16);
  int blocks = (tiles + 7) / 8;
  gemm_bf16_wmma<<<dim3(blocks, batches), 256, 0, s>>>(p);
}

extern "C" void kernel_launch(void* const* d_in, const int* in_sizes, int n_in,
                              void* d_out, int out_size, void* d_ws, size_t ws_size,
                              hipStream_t stream) {
  (void)in_sizes; (void)n_in; (void)out_size; (void)ws_size;
  const float* x     = (const float*)d_in[0];
  const float* ln1g  = (const float*)d_in[3];
  const float* ln1b  = (const float*)d_in[4];
  const float* eakw  = (const float*)d_in[5];
  const float* eakb  = (const float*)d_in[6];
  const float* eaqw  = (const float*)d_in[7];
  const float* eaqb  = (const float*)d_in[8];
  const float* eavw  = (const float*)d_in[9];
  const float* eavb  = (const float*)d_in[10];
  const float* earw  = (const float*)d_in[11];
  const float* earb  = (const float*)d_in[12];
  const float* ln2g  = (const float*)d_in[13];
  const float* ln2b  = (const float*)d_in[14];
  const float* m1w1  = (const float*)d_in[15];
  const float* m1b1  = (const float*)d_in[16];
  const float* m1dww = (const float*)d_in[17];
  const float* m1dwb = (const float*)d_in[18];
  const float* m1lg  = (const float*)d_in[19];
  const float* m1lb  = (const float*)d_in[20];
  const float* m1w2  = (const float*)d_in[21];
  const float* m1b2  = (const float*)d_in[22];
  const float* ln3g  = (const float*)d_in[23];
  const float* ln3b  = (const float*)d_in[24];
  const float* catemp= (const float*)d_in[25];
  const float* caqkvw= (const float*)d_in[26];
  const float* capw  = (const float*)d_in[27];
  const float* capb  = (const float*)d_in[28];
  const float* ln4g  = (const float*)d_in[29];
  const float* ln4b  = (const float*)d_in[30];
  const float* m2w1  = (const float*)d_in[31];
  const float* m2b1  = (const float*)d_in[32];
  const float* m2dww = (const float*)d_in[33];
  const float* m2dwb = (const float*)d_in[34];
  const float* m2lg  = (const float*)d_in[35];
  const float* m2lb  = (const float*)d_in[36];
  const float* m2w2  = (const float*)d_in[37];
  const float* m2b2  = (const float*)d_in[38];

  const long N3 = 3136;            // H*W
  const long BN = 8 * N3;          // 25088 rows

  // ---- workspace bump allocator (256B aligned) ----
  char* ws = (char*)d_ws;
  size_t off = 0;
  auto alloc = [&](size_t bytes) -> void* {
    void* p = ws + off;
    off = (off + bytes + 255) & ~(size_t)255;
    return p;
  };
  bf16_t* wk   = (bf16_t*)alloc(65536 * 2);
  bf16_t* wq   = (bf16_t*)alloc(65536 * 2);
  bf16_t* wv   = (bf16_t*)alloc(65536 * 2);
  bf16_t* wr   = (bf16_t*)alloc(65536 * 2);
  bf16_t* w11  = (bf16_t*)alloc(262144 * 2);
  bf16_t* w12  = (bf16_t*)alloc(262144 * 2);
  bf16_t* wqkv = (bf16_t*)alloc(196608 * 2);
  bf16_t* wp   = (bf16_t*)alloc(65536 * 2);
  bf16_t* w21  = (bf16_t*)alloc(262144 * 2);
  bf16_t* w22  = (bf16_t*)alloc(262144 * 2);
  bf16_t* xn   = (bf16_t*)alloc(BN * 256 * 2);
  char*   fslot  = (char*)alloc((size_t)BN * 1024 * 4);   // 102,760,448 B
  char*   axslot = (char*)alloc((size_t)BN * 1024 * 2);   //  51,380,224 B
  bf16_t* slotA  = (bf16_t*)alloc((size_t)8 * 256 * N3 * 2);
  bf16_t* slotB  = (bf16_t*)alloc((size_t)8 * 256 * N3 * 2);
  bf16_t* ctxT   = (bf16_t*)alloc((size_t)8 * 256 * 256 * 2);
  float*  addA   = (float*)alloc((size_t)BN * 256 * 4);   // add1 then add3
  float*  addB   = (float*)alloc((size_t)BN * 256 * 4);   // add2
  bf16_t* attn   = (bf16_t*)alloc((size_t)8 * 8 * 32 * 32 * 2);

  // overlapped views inside the big slots (lifetimes are disjoint):
  float*  kT_pre = (float*)fslot;                          // phase1
  float*  q_pre  = (float*)(fslot + 25690112);             // phase1
  bf16_t* qb     = (bf16_t*)(fslot + 51380224);            // phase1
  float*  fbuf   = (float*)fslot;                          // phase2/4
  float*  qkvT   = (float*)fslot;                          // phase3 (77,070,336 B)
  bf16_t* qk_b   = (bf16_t*)(fslot + 77070336);            // phase3 (25,690,112 B)
  bf16_t* axb    = (bf16_t*)axslot;                        // phase2/4
  bf16_t* qkv_nb = (bf16_t*)axslot;                        // phase3 (38,535,168 B)
  bf16_t* kT_b     = slotA;                                // phase1
  bf16_t* chan_out = slotA;                                // phase3
  bf16_t* vT_b = slotB;                                    // phase1 (then attb)
  bf16_t* attb = slotB;
  float* add1 = addA;
  float* add3 = addA;
  float* add2 = addB;

  auto cvt = [&](const float* src, bf16_t* dst, long n) {
    cvt_f32_bf16<<<dim3((unsigned)((n + 255) / 256)), 256, 0, stream>>>(src, dst, n);
  };
  cvt(eakw, wk, 65536);   cvt(eaqw, wq, 65536);   cvt(eavw, wv, 65536);
  cvt(earw, wr, 65536);   cvt(m1w1, w11, 262144); cvt(m1w2, w12, 262144);
  cvt(caqkvw, wqkv, 196608); cvt(capw, wp, 65536);
  cvt(m2w1, w21, 262144); cvt(m2w2, w22, 262144);

  GemmP g;
  // ---------------- Phase 1: efficient attention ----------------
  ln_rows256<<<3136, 256, 0, stream>>>(x, ln1g, ln1b, xn, BN);
  // k_pre (transposed f32 [B,256,N])
  g = GemmP{xn, N3 * 256, 256, wk, 0, 256, eakb, nullptr, 0,
            kT_pre, 256 * N3, 1, (int)N3, nullptr, 0, 0, 0, (int)N3, 256, 256};
  run_gemm(stream, g, 8);
  // q_pre (normal f32)
  g = GemmP{xn, N3 * 256, 256, wq, 0, 256, eaqb, nullptr, 0,
            q_pre, N3 * 256, 0, 256, nullptr, 0, 0, 0, (int)N3, 256, 256};
  run_gemm(stream, g, 8);
  // v (transposed bf16 [B,256,N])
  g = GemmP{xn, N3 * 256, 256, wv, 0, 256, eavb, nullptr, 0,
            nullptr, 0, 0, 0, vT_b, 256 * N3, 1, (int)N3, (int)N3, 256, 256};
  run_gemm(stream, g, 8);
  softmax_len<<<2048, 256, 0, stream>>>(kT_pre, kT_b, (int)N3);       // softmax over N
  softmax_rows256<<<3136, 256, 0, stream>>>(q_pre, qb, BN);           // softmax over C
  // ctxT[vc,kc] = sum_n vT[vc,n]*kT[kc,n]  (= ctx^T)
  g = GemmP{vT_b, 256 * N3, (int)N3, kT_b, 256 * N3, (int)N3, nullptr, nullptr, 0,
            nullptr, 0, 0, 0, ctxT, 65536, 0, 256, 256, 256, (int)N3};
  run_gemm(stream, g, 8);
  // att[n,vc] = sum_k q[n,k]*ctxT[vc,k]
  g = GemmP{qb, N3 * 256, 256, ctxT, 65536, 256, nullptr, nullptr, 0,
            nullptr, 0, 0, 0, attb, N3 * 256, 0, 256, (int)N3, 256, 256};
  run_gemm(stream, g, 8);
  // add1 = x + att @ rw^T + rb
  g = GemmP{attb, N3 * 256, 256, wr, 0, 256, earb, x, N3 * 256,
            add1, N3 * 256, 0, 256, nullptr, 0, 0, 0, (int)N3, 256, 256};
  run_gemm(stream, g, 8);

  // ---------------- Phase 2: MixFFN 1 ----------------
  ln_rows256<<<3136, 256, 0, stream>>>(add1, ln2g, ln2b, xn, BN);
  g = GemmP{xn, N3 * 256, 256, w11, 0, 256, m1b1, nullptr, 0,
            fbuf, N3 * 1024, 0, 1024, nullptr, 0, 0, 0, (int)N3, 1024, 256};
  run_gemm(stream, g, 8);
  dwconv_ln_gelu<<<dim3(3136, 8), 256, 0, stream>>>(fbuf, m1dww, m1dwb, m1lg, m1lb, axb);
  g = GemmP{axb, N3 * 1024, 1024, w12, 0, 1024, m1b2, add1, N3 * 256,
            add2, N3 * 256, 0, 256, nullptr, 0, 0, 0, (int)N3, 256, 1024};
  run_gemm(stream, g, 8);

  // ---------------- Phase 3: channel attention ----------------
  ln_rows256<<<3136, 256, 0, stream>>>(add2, ln3g, ln3b, xn, BN);
  // qkv: transposed f32 [B,768,N] + normal bf16 [B,N,768]
  g = GemmP{xn, N3 * 256, 256, wqkv, 0, 256, nullptr, nullptr, 0,
            qkvT, 768 * N3, 1, (int)N3, qkv_nb, N3 * 768, 0, 768, (int)N3, 768, 256};
  run_gemm(stream, g, 8);
  l2norm_rows<<<4096, 256, 0, stream>>>(qkvT, qk_b);
  chan_attn_scores<<<dim3(8, 8), 128, 0, stream>>>(qk_b, catemp, attn);
  chan_attn_out<<<dim3(49, 8, 8), 256, 0, stream>>>(qkv_nb, attn, chan_out);
  // add3 = add2 + chan_out @ pw^T + pb
  g = GemmP{chan_out, N3 * 256, 256, wp, 0, 256, capb, add2, N3 * 256,
            add3, N3 * 256, 0, 256, nullptr, 0, 0, 0, (int)N3, 256, 256};
  run_gemm(stream, g, 8);

  // ---------------- Phase 4: MixFFN 2 ----------------
  ln_rows256<<<3136, 256, 0, stream>>>(add3, ln4g, ln4b, xn, BN);
  g = GemmP{xn, N3 * 256, 256, w21, 0, 256, m2b1, nullptr, 0,
            fbuf, N3 * 1024, 0, 1024, nullptr, 0, 0, 0, (int)N3, 1024, 256};
  run_gemm(stream, g, 8);
  dwconv_ln_gelu<<<dim3(3136, 8), 256, 0, stream>>>(fbuf, m2dww, m2dwb, m2lg, m2lb, axb);
  g = GemmP{axb, N3 * 1024, 1024, w22, 0, 1024, m2b2, add3, N3 * 256,
            (float*)d_out, N3 * 256, 0, 256, nullptr, 0, 0, 0, (int)N3, 256, 1024};
  run_gemm(stream, g, 8);
}
